// GBEncoderBlock_28484223107495
// MI455X (gfx1250) — matrix operand order repaired and checked by reading, not compiled, not run on hardware
//
#include <hip/hip_runtime.h>

#define B_  8
#define L_  1024
#define H_  512
#define NH_ 8
#define DK_ 64
#define BL_ (B_ * L_)          // 8192 rows
#define EPS_ 1e-6f

typedef __attribute__((ext_vector_type(16))) _Float16 v16h;
typedef __attribute__((ext_vector_type(8)))  _Float16 v8h;
typedef __attribute__((ext_vector_type(8)))  float    v8f;

static __device__ __forceinline__ v16h cat16(v8h lo, v8h hi) {
  return __builtin_shufflevector(lo, hi, 0,1,2,3,4,5,6,7,8,9,10,11,12,13,14,15);
}
// A fragment (16x32 f16): lane holds K {ab..ab+7, ab+16..ab+23}, ab = (lane&16)?8:0.
static __device__ __forceinline__ v16h loadA(const _Float16* p) {
  v8h lo = *(const v8h*)p;
  v8h hi = *(const v8h*)(p + 16);
  return cat16(lo, hi);
}
// B fragment (32x16 f16): lane holds K {bb..bb+7, bb+16..bb+23} at its column, bb=(lane&16)?8:0.
static __device__ __forceinline__ v16h loadB(const _Float16* p) {
  v8h lo = *(const v8h*)p;
  v8h hi = *(const v8h*)(p + 16);
  return cat16(lo, hi);
}
static __device__ __forceinline__ v8f wmma16(v16h a, v16h b, v8f c) {
  return __builtin_amdgcn_wmma_f32_16x16x32_f16(false, a, false, b, (short)0, c, false, false);
}

// ---- CDNA5 async global->LDS copy (ASYNCcnt path, ISA 15.18.3 op 98) ----
static __device__ __forceinline__ void async_cp16(unsigned lds_off, const _Float16* gptr) {
  asm volatile("global_load_async_to_lds_b128 %0, %1, off"
               :: "v"(lds_off), "v"((unsigned long long)(uintptr_t)gptr)
               : "memory");
}
static __device__ __forceinline__ void async_wait0() {
  asm volatile("s_wait_asynccnt 0x0" ::: "memory");
}

// ---------------- LDS-blocked WMMA GEMM: C(MxN) = A(MxK,f16) * W(NxK,f16)^T --------
// Block: 256 thr = 8 waves, 128x128 output tile; wave = 32x64 (2x4 subtiles, 8 wmma/K-step).
enum { EP_QK = 0, EP_VT = 1, EP_PW = 2, EP_RES = 3, EP_RELU16 = 4 };

template <int EP>
__global__ __launch_bounds__(256)
void wmma_gemm(const _Float16* __restrict__ A, const _Float16* __restrict__ W,
               const float* __restrict__ bias, const float* __restrict__ resid,
               float* __restrict__ outF, _Float16* __restrict__ outH,
               int M, int N, int K) {
  __shared__ __align__(16) _Float16 Al[128 * 40];   // 128 rows x 32 K (+8 pad)
  __shared__ __align__(16) _Float16 Bl[128 * 40];
  const int tid  = threadIdx.x;
  const int lane = tid & 31;
  const int wave = tid >> 5;
  const int nb   = N >> 7;
  const int bm   = blockIdx.x / nb;
  const int bn   = blockIdx.x - bm * nb;
  const int m0b  = bm << 7, n0b = bn << 7;
  const int wm   = wave >> 1, wn = wave & 1;        // 4 x 2 wave grid

  const int  n  = lane & 15;
  const bool hi = (lane & 16) != 0;
  const int  ab = hi ? 8 : 0;
  const int  bb = hi ? 8 : 0;

  const unsigned aoff = (unsigned)(uintptr_t)Al;
  const unsigned boff = (unsigned)(uintptr_t)Bl;
  const int r0 = tid >> 2, q0 = tid & 3;            // staging chunk: row r0 / r0+64, quad q0

  v8f acc[2][4] = {};
  for (int k0 = 0; k0 < K; k0 += 32) {
    // stage A(128x32) and W(128x32) tiles via async DMA to LDS
    async_cp16(aoff + (unsigned)(r0 * 80 + q0 * 16),        A + (size_t)(m0b + r0) * K + k0 + q0 * 8);
    async_cp16(aoff + (unsigned)((r0 + 64) * 80 + q0 * 16), A + (size_t)(m0b + r0 + 64) * K + k0 + q0 * 8);
    async_cp16(boff + (unsigned)(r0 * 80 + q0 * 16),        W + (size_t)(n0b + r0) * K + k0 + q0 * 8);
    async_cp16(boff + (unsigned)((r0 + 64) * 80 + q0 * 16), W + (size_t)(n0b + r0 + 64) * K + k0 + q0 * 8);
    __builtin_prefetch(A + (size_t)(m0b + r0) * K + k0 + 32, 0, 1);
    async_wait0();
    __syncthreads();

    v16h af0 = loadA(Al + (wm * 32 + n) * 40 + ab);
    v16h af1 = loadA(Al + (wm * 32 + 16 + n) * 40 + ab);
    v16h bf0 = loadB(Bl + (wn * 64 + n) * 40 + bb);
    v16h bf1 = loadB(Bl + (wn * 64 + 16 + n) * 40 + bb);
    v16h bf2 = loadB(Bl + (wn * 64 + 32 + n) * 40 + bb);
    v16h bf3 = loadB(Bl + (wn * 64 + 48 + n) * 40 + bb);
    acc[0][0] = wmma16(af0, bf0, acc[0][0]);
    acc[0][1] = wmma16(af0, bf1, acc[0][1]);
    acc[0][2] = wmma16(af0, bf2, acc[0][2]);
    acc[0][3] = wmma16(af0, bf3, acc[0][3]);
    acc[1][0] = wmma16(af1, bf0, acc[1][0]);
    acc[1][1] = wmma16(af1, bf1, acc[1][1]);
    acc[1][2] = wmma16(af1, bf2, acc[1][2]);
    acc[1][3] = wmma16(af1, bf3, acc[1][3]);
    __syncthreads();
  }

  const int mB = hi ? 8 : 0;
#pragma unroll
  for (int ts = 0; ts < 2; ++ts)
#pragma unroll
    for (int tu = 0; tu < 4; ++tu)
#pragma unroll
      for (int r = 0; r < 8; ++r) {
        const int row = m0b + wm * 32 + ts * 16 + mB + r;
        const int col = n0b + wn * 64 + tu * 16 + n;
        float v = acc[ts][tu][r];
        if constexpr (EP == EP_QK) {          // [B,L,(h,k)] -> [b,h,l,k] f16
          const int b = row >> 10, l = row & 1023;
          const int h = col >> 6,  k = col & 63;
          outH[((size_t)((b * 8 + h) * 1024 + l)) * 64 + k] = (_Float16)v;
        } else if constexpr (EP == EP_VT) {   // -> [b,h,k,l] f16 (V transposed)
          const int b = row >> 10, l = row & 1023;
          const int h = col >> 6,  k = col & 63;
          outH[((size_t)((b * 8 + h) * 64 + k)) * 1024 + l] = (_Float16)v;
        } else if constexpr (EP == EP_PW) {   // resid += relu(acc + bias)
          float t = v + bias[col];
          t = t > 0.f ? t : 0.f;
          const size_t idx = (size_t)row * N + col;
          outF[idx] = resid[idx] + t;
        } else if constexpr (EP == EP_RES) {  // outF = resid + acc + bias
          const size_t idx = (size_t)row * N + col;
          outF[idx] = resid[idx] + v + bias[col];
        } else {                              // EP_RELU16: f16 relu(acc + bias)
          float t = v + bias[col];
          outH[(size_t)row * N + col] = (_Float16)(t > 0.f ? t : 0.f);
        }
      }
}

// ---------------- flash attention: block = one (b,h), 8 waves x 16 query rows -------
// K/V chunks staged in LDS once per block (8x less HBM traffic than per-wave streams).
__global__ __launch_bounds__(256)
void flash_attn(const _Float16* __restrict__ Q, const _Float16* __restrict__ Kb,
                const _Float16* __restrict__ Vt, const unsigned char* __restrict__ mask,
                _Float16* __restrict__ O) {
  __shared__ __align__(16) _Float16 Kl[32 * 72];    // 32 keys x 64 dk (+8 pad)
  __shared__ __align__(16) _Float16 Vl[64 * 40];    // 64 dk x 32 keys (+8 pad)
  __shared__ __align__(16) _Float16 Pl[8][16 * 40]; // per-wave P transpose staging
  const int tid  = threadIdx.x;
  const int lane = tid & 31;
  const int wave = tid >> 5;
  const int bh = blockIdx.x >> 3;                   // b*8 + h
  const int b  = bh >> 3;
  const int l0 = (blockIdx.x & 7) * 128 + wave * 16;

  const int  n  = lane & 15;
  const bool hi = (lane & 16) != 0;
  const int  ab = hi ? 8 : 0;
  const int  bb = hi ? 8 : 0;
  const int  mB = hi ? 8 : 0;
  const float NEGINF = -__builtin_inff();

  const unsigned koff = (unsigned)(uintptr_t)Kl;
  const unsigned voff = (unsigned)(uintptr_t)Vl;
  const int kr = tid >> 3, kq = tid & 7;            // K staging chunk (32x8 quads)
  const int vr = tid >> 2, vq = tid & 3;            // V staging chunk (64x4 quads)

  const _Float16* qp = Q + ((size_t)bh * 1024 + (l0 + n)) * 64 + ab;
  v16h qa0 = loadA(qp);        // k 0..31
  v16h qa1 = loadA(qp + 32);   // k 32..63

  float mSt[8], lSt[8];
  v8f Og[4] = {};
#pragma unroll
  for (int r = 0; r < 8; ++r) { mSt[r] = NEGINF; lSt[r] = 0.f; }

  for (int m0 = 0; m0 < 1024; m0 += 32) {
    // cooperatively stage K(32x64) and Vt(64x32) chunks
    async_cp16(koff + (unsigned)(kr * 144 + kq * 16),
               Kb + ((size_t)bh * 1024 + m0 + kr) * 64 + kq * 8);
    async_cp16(voff + (unsigned)(vr * 80 + vq * 16),
               Vt + ((size_t)bh * 64 + vr) * 1024 + m0 + vq * 8);
    async_wait0();
    __syncthreads();

    // two 16x16 score tiles from LDS
    const _Float16* kp0 = Kl + n * 72 + bb;
    const _Float16* kp1 = Kl + (16 + n) * 72 + bb;
    v8f s0 = {}, s1 = {};
    s0 = wmma16(qa0, loadB(kp0), s0);
    s0 = wmma16(qa1, loadB(kp0 + 32), s0);
    s1 = wmma16(qa0, loadB(kp1), s1);
    s1 = wmma16(qa1, loadB(kp1 + 32), s1);

    const bool mk0 = mask[b * 1024 + m0 + n] != 0;        // lane-uniform per column
    const bool mk1 = mask[b * 1024 + m0 + 16 + n] != 0;

    float p0[8], p1[8], cor[8];
#pragma unroll
    for (int r = 0; r < 8; ++r) {
      p0[r] = mk0 ? NEGINF : s0[r] * 0.125f;   // 1/sqrt(64)
      p1[r] = mk1 ? NEGINF : s1[r] * 0.125f;
    }
#pragma unroll
    for (int r = 0; r < 8; ++r) {
      float rm = fmaxf(p0[r], p1[r]);
#pragma unroll
      for (int off = 8; off >= 1; off >>= 1) rm = fmaxf(rm, __shfl_xor(rm, off, 16));
      float mnew = fmaxf(mSt[r], rm);
      float c, e0, e1;
      if (mnew == NEGINF) { c = 1.f; e0 = 0.f; e1 = 0.f; }
      else {
        c  = __expf(mSt[r] - mnew);   // exp(-inf)=0 handles first iteration
        e0 = __expf(p0[r] - mnew);
        e1 = __expf(p1[r] - mnew);
      }
      float rs = e0 + e1;
#pragma unroll
      for (int off = 8; off >= 1; off >>= 1) rs += __shfl_xor(rs, off, 16);
      lSt[r] = lSt[r] * c + rs;
      mSt[r] = mnew;
      cor[r] = c;
      p0[r] = e0; p1[r] = e1;
    }
#pragma unroll
    for (int g = 0; g < 4; ++g)
#pragma unroll
      for (int r = 0; r < 8; ++r) Og[g][r] *= cor[r];

    // C-layout P -> LDS -> A-layout fragment (16x32)
#pragma unroll
    for (int r = 0; r < 8; ++r) {
      Pl[wave][(mB + r) * 40 + n]      = (_Float16)p0[r];
      Pl[wave][(mB + r) * 40 + 16 + n] = (_Float16)p1[r];
    }
    __syncthreads();
    v16h pa = loadA(&Pl[wave][n * 40 + ab]);

    // O(16x64) += P(16x32) x V(32x64), V fragments from LDS
#pragma unroll
    for (int g = 0; g < 4; ++g) {
      const _Float16* vp = Vl + (g * 16 + n) * 40 + bb;
      Og[g] = wmma16(pa, loadB(vp), Og[g]);
    }
    __syncthreads();   // protect Kl/Vl/Pl before next iteration's staging
  }

#pragma unroll
  for (int g = 0; g < 4; ++g)
#pragma unroll
    for (int r = 0; r < 8; ++r) {
      float d = lSt[r];
      float v = (d > 0.f) ? Og[g][r] / d : 0.f;   // fully-masked rows -> 0 (NaN fix)
      const int row = l0 + mB + r;
      const int col = (bh & 7) * 64 + g * 16 + n;
      O[((size_t)b * 1024 + row) * 512 + col] = (_Float16)v;
    }
}

// ---------------- LayerNorm over last dim (H=512), unbiased std, f16 out -----------
__global__ __launch_bounds__(128)
void ln_rows(const float* __restrict__ x, const float* __restrict__ gamma,
             const float* __restrict__ beta, _Float16* __restrict__ out) {
  __shared__ float red[128];
  const int row = blockIdx.x;
  const int tid = threadIdx.x;
  const float* xr = x + (size_t)row * 512;
  float v[4];
  float s = 0.f;
#pragma unroll
  for (int j = 0; j < 4; ++j) { v[j] = xr[tid + 128 * j]; s += v[j]; }
  red[tid] = s; __syncthreads();
  for (int o = 64; o > 0; o >>= 1) { if (tid < o) red[tid] += red[tid + o]; __syncthreads(); }
  const float mean = red[0] * (1.f / 512.f);
  __syncthreads();
  float ss = 0.f;
#pragma unroll
  for (int j = 0; j < 4; ++j) { float d = v[j] - mean; ss += d * d; }
  red[tid] = ss; __syncthreads();
  for (int o = 64; o > 0; o >>= 1) { if (tid < o) red[tid] += red[tid + o]; __syncthreads(); }
  const float stdv = sqrtf(red[0] * (1.f / 511.f));
  const float inv  = 1.f / (stdv + EPS_);
#pragma unroll
  for (int j = 0; j < 4; ++j) {
    int c = tid + 128 * j;
    out[(size_t)row * 512 + c] = (_Float16)(gamma[c] * (v[j] - mean) * inv + beta[c]);
  }
}

// ---------------- depthwise conv over L, K=7, pad 3 ----------------
__global__ __launch_bounds__(256)
void dwconv(const _Float16* __restrict__ in, const float* __restrict__ w,
            const float* __restrict__ bias, _Float16* __restrict__ out) {
  int i = blockIdx.x * blockDim.x + threadIdx.x;
  if (i >= B_ * L_ * H_) return;
  int c = i & 511, l = (i >> 9) & 1023, b = i >> 19;
  float acc = bias[c];
#pragma unroll
  for (int j = 0; j < 7; ++j) {
    int ll = l + j - 3;
    if (ll >= 0 && ll < 1024)
      acc += w[c * 7 + j] * (float)in[((size_t)b << 19) + ((size_t)ll << 9) + c];
  }
  out[i] = (_Float16)acc;
}

// ---------------- misc small kernels ----------------
__global__ void pos_add(const float* __restrict__ x, const float* __restrict__ pos,
                        float* __restrict__ resid) {
  int i = blockIdx.x * blockDim.x + threadIdx.x;
  if (i >= B_ * L_ * H_) return;
  int c = i & 511, l = (i >> 9) & 1023;
  resid[i] = x[i] + pos[l * 512 + c];
}

__global__ void cvt_f32_f16(const float* __restrict__ in, _Float16* __restrict__ out, int n) {
  int i = blockIdx.x * blockDim.x + threadIdx.x;
  if (i < n) out[i] = (_Float16)in[i];
}

// w_q/k/v [h,d,k] f32 -> [(h*64+k), d] f16  (so the GEMM B-operand is contiguous-K)
__global__ void cvt_qkvT(const float* __restrict__ w, _Float16* __restrict__ out) {
  int idx = blockIdx.x * blockDim.x + threadIdx.x;
  if (idx >= 512 * 512) return;
  int o = idx >> 9, d = idx & 511;
  int h = o >> 6, k = o & 63;
  out[idx] = (_Float16)w[((h << 9) + d) * 64 + k];
}

// ---------------- launch ----------------
extern "C" void kernel_launch(void* const* d_in, const int* in_sizes, int n_in,
                              void* d_out, int out_size, void* d_ws, size_t ws_size,
                              hipStream_t stream) {
  (void)in_sizes; (void)n_in; (void)out_size; (void)ws_size;
  const float* x          = (const float*)d_in[0];
  const unsigned char* xm = (const unsigned char*)d_in[1];   // jax bool = 1 byte
  const float* pos        = (const float*)d_in[2];
  const float* cnn_gamma  = (const float*)d_in[3];
  const float* cnn_beta   = (const float*)d_in[4];
  const float* cnn_dw_w   = (const float*)d_in[5];
  const float* cnn_dw_b   = (const float*)d_in[6];
  const float* cnn_pw_w   = (const float*)d_in[7];
  const float* cnn_pw_b   = (const float*)d_in[8];
  const float* attn_gamma = (const float*)d_in[9];
  const float* attn_beta  = (const float*)d_in[10];
  const float* w_qs       = (const float*)d_in[11];
  const float* w_ks       = (const float*)d_in[12];
  const float* w_vs       = (const float*)d_in[13];
  const float* proj_w     = (const float*)d_in[14];
  const float* proj_b     = (const float*)d_in[15];
  const float* ffn_gamma  = (const float*)d_in[16];
  const float* ffn_beta   = (const float*)d_in[17];
  const float* ffn_w1     = (const float*)d_in[18];
  const float* ffn_b1     = (const float*)d_in[19];
  const float* ffn_w2     = (const float*)d_in[20];
  const float* ffn_b2     = (const float*)d_in[21];

  // workspace bump allocator (~76 MB total)
  char* p = (char*)d_ws;
  auto alloc = [&](size_t bytes) {
    char* r = p; p += (bytes + 255) & ~(size_t)255; return r;
  };
  const int NE = B_ * L_ * H_;                      // 4,194,304
  float*    resid  = (float*)   alloc(sizeof(float) * NE);
  _Float16* ln16   = (_Float16*)alloc(2ull * NE);
  _Float16* bufB   = (_Float16*)alloc(2ull * NE * 4);   // dw / q,k,vt / ffn1
  _Float16* attno  = (_Float16*)alloc(2ull * NE);
  _Float16* wpw16  = (_Float16*)alloc(2ull * 4 * H_ * H_);
  _Float16* wq16   = (_Float16*)alloc(2ull * H_ * H_);
  _Float16* wk16   = (_Float16*)alloc(2ull * H_ * H_);
  _Float16* wv16   = (_Float16*)alloc(2ull * H_ * H_);
  _Float16* wproj  = (_Float16*)alloc(2ull * H_ * H_);
  _Float16* w116   = (_Float16*)alloc(2ull * 4 * H_ * H_);
  _Float16* w216   = (_Float16*)alloc(2ull * 4 * H_ * H_);

  // weight conversions (f32 -> f16, [out,in] layout)
  cvt_f32_f16<<<(4 * H_ * H_ + 255) / 256, 256, 0, stream>>>(cnn_pw_w, wpw16, 4 * H_ * H_);
  cvt_f32_f16<<<(H_ * H_ + 255) / 256, 256, 0, stream>>>(proj_w, wproj, H_ * H_);
  cvt_f32_f16<<<(4 * H_ * H_ + 255) / 256, 256, 0, stream>>>(ffn_w1, w116, 4 * H_ * H_);
  cvt_f32_f16<<<(4 * H_ * H_ + 255) / 256, 256, 0, stream>>>(ffn_w2, w216, 4 * H_ * H_);
  cvt_qkvT<<<(H_ * H_ + 255) / 256, 256, 0, stream>>>(w_qs, wq16);
  cvt_qkvT<<<(H_ * H_ + 255) / 256, 256, 0, stream>>>(w_ks, wk16);
  cvt_qkvT<<<(H_ * H_ + 255) / 256, 256, 0, stream>>>(w_vs, wv16);

  pos_add<<<(NE + 255) / 256, 256, 0, stream>>>(x, pos, resid);

  auto ggrid = [](int M, int N) { return (M >> 7) * (N >> 7); };

  // ---- 4 conv layers ----
  for (int i = 0; i < 4; ++i) {
    ln_rows<<<BL_, 128, 0, stream>>>(resid, cnn_gamma + i * H_, cnn_beta + i * H_, ln16);
    dwconv<<<(NE + 255) / 256, 256, 0, stream>>>(ln16, cnn_dw_w + i * H_ * 7,
                                                 cnn_dw_b + i * H_, bufB);
    wmma_gemm<EP_PW><<<ggrid(BL_, H_), 256, 0, stream>>>(
        bufB, wpw16 + i * H_ * H_, cnn_pw_b + i * H_, resid, resid, nullptr, BL_, H_, H_);
  }

  // ---- attention ----
  ln_rows<<<BL_, 128, 0, stream>>>(resid, attn_gamma, attn_beta, ln16);
  _Float16* q16 = bufB;
  _Float16* k16 = bufB + (size_t)NE;
  _Float16* vt16 = bufB + 2 * (size_t)NE;
  wmma_gemm<EP_QK><<<ggrid(BL_, H_), 256, 0, stream>>>(
      ln16, wq16, nullptr, nullptr, nullptr, q16, BL_, H_, H_);
  wmma_gemm<EP_QK><<<ggrid(BL_, H_), 256, 0, stream>>>(
      ln16, wk16, nullptr, nullptr, nullptr, k16, BL_, H_, H_);
  wmma_gemm<EP_VT><<<ggrid(BL_, H_), 256, 0, stream>>>(
      ln16, wv16, nullptr, nullptr, nullptr, vt16, BL_, H_, H_);
  flash_attn<<<512, 256, 0, stream>>>(q16, k16, vt16, xm, attno);
  wmma_gemm<EP_RES><<<ggrid(BL_, H_), 256, 0, stream>>>(
      attno, wproj, proj_b, resid, resid, nullptr, BL_, H_, H_);

  // ---- FFN ----
  ln_rows<<<BL_, 128, 0, stream>>>(resid, ffn_gamma, ffn_beta, ln16);
  wmma_gemm<EP_RELU16><<<ggrid(BL_, 4 * H_), 256, 0, stream>>>(
      ln16, w116, ffn_b1, nullptr, nullptr, bufB, BL_, 4 * H_, H_);
  wmma_gemm<EP_RES><<<ggrid(BL_, H_), 256, 0, stream>>>(
      bufB, w216, ffn_b2, resid, (float*)d_out, nullptr, BL_, H_, 4 * H_);
}
